// M_833223656256
// MI455X (gfx1250) — matrix operand order repaired and checked by reading, not compiled, not run on hardware
//
#include <hip/hip_runtime.h>
#include <hip/hip_bf16.h>
#include <math.h>

// ---------------------------------------------------------------------------
// Problem constants (fixed by the reference): M=32*512, K=768, N=3072
// ---------------------------------------------------------------------------
#define M_DIM 16384
#define K_DIM 768
#define N_DIM 3072

typedef __attribute__((ext_vector_type(16))) __bf16 v16bf;
typedef __attribute__((ext_vector_type(8)))  float  v8f;
typedef __attribute__((ext_vector_type(4)))  int    v4i;

// pointer-to-AS(1)-int4 and pointer-to-AS(3)-int4 (builtin's exact param types)
typedef __attribute__((address_space(1))) v4i* as1_v4i_p;
typedef __attribute__((address_space(3))) v4i* as3_v4i_p;

// ---------------------------------------------------------------------------
// gfx1250 async global->LDS copy (ASYNCcnt path), with safe fallback.
// ---------------------------------------------------------------------------
#if defined(__gfx1250__) && __has_builtin(__builtin_amdgcn_global_load_async_to_lds_b128)
#define HAS_ASYNC_LDS 1
#else
#define HAS_ASYNC_LDS 0
#endif

__device__ __forceinline__ void copy16_g2l(const __bf16* g, __bf16* l)
{
#if HAS_ASYNC_LDS
    __builtin_amdgcn_global_load_async_to_lds_b128(
        (as1_v4i_p)(g), (as3_v4i_p)(l), /*imm offset*/0, /*cpol*/0);
#else
    *(uint4*)l = *(const uint4*)g;
#endif
}

__device__ __forceinline__ void wait_async_copies()
{
#if HAS_ASYNC_LDS
#if __has_builtin(__builtin_amdgcn_s_wait_asynccnt)
    __builtin_amdgcn_s_wait_asynccnt(0);
#else
    asm volatile("s_wait_asynccnt 0x0" ::: "memory");
#endif
#endif
}

// ---------------------------------------------------------------------------
// Kernel 1: LayerNorm (fp32 stats) -> bf16 normalized activations.
// One wave32 per row of 768; 24 elements per lane; wave shuffle reduction.
// ---------------------------------------------------------------------------
__global__ __launch_bounds__(256)
void ln_bf16_kernel(const float* __restrict__ x,
                    const float* __restrict__ gamma,
                    const float* __restrict__ beta,
                    __bf16* __restrict__ xn)
{
    const int wave = threadIdx.x >> 5;
    const int lane = threadIdx.x & 31;
    const int row  = blockIdx.x * 8 + wave;   // grid = M/8 blocks

    const float* xr = x + (size_t)row * K_DIM;

    float vals[24];
    float sum = 0.0f, sq = 0.0f;
#pragma unroll
    for (int i = 0; i < 24; ++i) {
        float v = xr[lane + i * 32];
        vals[i] = v;
        sum += v;
        sq  += v * v;
    }
#pragma unroll
    for (int off = 16; off >= 1; off >>= 1) {
        sum += __shfl_xor(sum, off, 32);
        sq  += __shfl_xor(sq,  off, 32);
    }
    const float inv_k = 1.0f / (float)K_DIM;
    const float mean  = sum * inv_k;
    const float var   = sq * inv_k - mean * mean;
    const float rstd  = rsqrtf(var + 1e-12f);

    __bf16* orow = xn + (size_t)row * K_DIM;
#pragma unroll
    for (int i = 0; i < 24; ++i) {
        const int k = lane + i * 32;
        float v = (vals[i] - mean) * rstd * gamma[k] + beta[k];
        orow[k] = (__bf16)v;
    }
}

// ---------------------------------------------------------------------------
// Kernel 2: W fp32 [K][N] -> bf16 transposed Wt [N][K], LDS-tiled 32x32.
// Makes the GEMM's B-tile k-contiguous so staging needs no scalar transpose.
// ---------------------------------------------------------------------------
__global__ __launch_bounds__(256)
void w_transpose_bf16_kernel(const float* __restrict__ W,
                             __bf16* __restrict__ Wt)
{
    __shared__ float tile[32][33];

    const int k0 = blockIdx.x * 32;     // K/32 = 24
    const int n0 = blockIdx.y * 32;     // N/32 = 96

    // coalesced read: 8 rows x 32 cols per pass, 4 passes
    const int r = threadIdx.x >> 5;
    const int c = threadIdx.x & 31;
#pragma unroll
    for (int i = 0; i < 4; ++i)
        tile[r + 8 * i][c] = W[(size_t)(k0 + r + 8 * i) * N_DIM + n0 + c];

    __syncthreads();

    // write Wt[n][k] as bf16, 4 contiguous k per thread (8-byte stores)
    const int n  = threadIdx.x >> 3;        // 0..31
    const int kc = (threadIdx.x & 7) * 4;   // 0..28
    union { __bf16 h[4]; uint2 u; } pk;
#pragma unroll
    for (int j = 0; j < 4; ++j)
        pk.h[j] = (__bf16)tile[kc + j][n];
    *(uint2*)(&Wt[(size_t)(n0 + n) * K_DIM + k0 + kc]) = pk.u;
}

// ---------------------------------------------------------------------------
// Kernel 3: bf16 WMMA GEMM (128x128 tile, BK=32) + bias + exact erf-GELU.
//   8 waves/block; each wave computes a 32x64 subtile = 2x4 wmma tiles.
//   Double-buffered LDS; both A and B tiles stage with async b128
//   global->LDS copies (ASYNCcnt) on gfx1250.
// ---------------------------------------------------------------------------
#define BM 128
#define BN 128
#define BK 32
#define LDT 48   // padded stride (bf16 elems): 96 B rows, 16 B-aligned

union FragU { uint4 u[2]; v16bf v; };

__global__ __launch_bounds__(256)
void gemm_bias_gelu_kernel(const __bf16* __restrict__ A,    // [M][K] bf16
                           const __bf16* __restrict__ Bt,   // [N][K] bf16
                           const float*  __restrict__ bias, // [N]
                           float* __restrict__ C)           // [M][N] fp32
{
    __shared__ __bf16 As[2][BM * LDT];
    __shared__ __bf16 Bs[2][BN * LDT];

    const int tid    = threadIdx.x;
    const int wave   = tid >> 5;
    const int lane   = tid & 31;
    const int lane16 = lane & 15;
    const int laneHi = lane >> 4;       // 0 | 1

    const int block_m = blockIdx.y * BM;
    const int block_n = blockIdx.x * BN;

    const int wm = wave >> 1;           // 0..3 : 32-row strip
    const int wn = wave & 1;            // 0..1 : 64-col strip

    v8f acc[2][4] = {};                 // fp32 accumulators

    // staging map (identical for A and B, both 128 x 32 k-contiguous):
    // thread -> (row = tid/2, 16 contiguous bf16 at k-chunk = (tid&1)*16)
    const int s_row = tid >> 1;
    const int s_off = (tid & 1) * 16;

    const __bf16* gA = A  + (size_t)(block_m + s_row) * K_DIM + s_off;
    const __bf16* gB = Bt + (size_t)(block_n + s_row) * K_DIM + s_off;
    __bf16* lA0 = &As[0][s_row * LDT + s_off];
    __bf16* lA1 = &As[1][s_row * LDT + s_off];
    __bf16* lB0 = &Bs[0][s_row * LDT + s_off];
    __bf16* lB1 = &Bs[1][s_row * LDT + s_off];

    const int NT = K_DIM / BK;          // 24 k-tiles

    // prologue: stage tile 0 into buffer 0
    copy16_g2l(gA, lA0);
    copy16_g2l(gA + 8, lA0 + 8);
    copy16_g2l(gB, lB0);
    copy16_g2l(gB + 8, lB0 + 8);
    wait_async_copies();
    __syncthreads();

    for (int t = 0; t < NT; ++t) {
        const int cur = t & 1;

        // stage tile t+1 into the other buffer while computing tile t
        if (t + 1 < NT) {
            const int ko = (t + 1) * BK;
            __bf16* la = cur ? lA0 : lA1;
            __bf16* lb = cur ? lB0 : lB1;
            copy16_g2l(gA + ko,     la);
            copy16_g2l(gA + ko + 8, la + 8);
            copy16_g2l(gB + ko,     lb);
            copy16_g2l(gB + ko + 8, lb + 8);
        }

        // ---- fragments from buffer `cur` ----
        // A (16x32 bf16): lane = row (lane%16); K halves {0..7,16..23}/{8..15,24..31}
        v16bf afrag[2];
#pragma unroll
        for (int mi = 0; mi < 2; ++mi) {
            const int row = wm * 32 + mi * 16 + lane16;
            const int k8  = laneHi * 8;
            FragU f;
            f.u[0] = *(const uint4*)(&As[cur][row * LDT + k8]);
            f.u[1] = *(const uint4*)(&As[cur][row * LDT + k8 + 16]);
            afrag[mi] = f.v;
        }
        // B (32x16 bf16): lane = col (lane%16); K halves {0..7,16..23}/{8..15,24..31}
        v16bf bfrag[4];
#pragma unroll
        for (int ni = 0; ni < 4; ++ni) {
            const int col = wn * 64 + ni * 16 + lane16;
            const int k16 = laneHi * 8;
            FragU f;
            f.u[0] = *(const uint4*)(&Bs[cur][col * LDT + k16]);
            f.u[1] = *(const uint4*)(&Bs[cur][col * LDT + k16 + 16]);
            bfrag[ni] = f.v;
        }

        // ---- 8 WMMAs: D = A x B + C ----
#pragma unroll
        for (int mi = 0; mi < 2; ++mi)
#pragma unroll
            for (int ni = 0; ni < 4; ++ni)
                acc[mi][ni] = __builtin_amdgcn_wmma_f32_16x16x32_bf16(
                    /*neg_a=*/false, afrag[mi],
                    /*neg_b=*/false, bfrag[ni],
                    /*c_mod=*/(short)0, acc[mi][ni],
                    /*reuse_a=*/false, /*reuse_b=*/false);

        // staged tile must have landed in LDS before anyone reads it next iter
        wait_async_copies();
        __syncthreads();
    }

    // ---- epilogue: bias + exact GELU, fp32 store ----
    // C/D layout: VGPR e holds row (e + 8*laneHi), col = lane%16.
#pragma unroll
    for (int mi = 0; mi < 2; ++mi) {
        const int base_m = block_m + wm * 32 + mi * 16 + laneHi * 8;
#pragma unroll
        for (int ni = 0; ni < 4; ++ni) {
            const int col = block_n + wn * 64 + ni * 16 + lane16;
            const float bb = bias[col];
#pragma unroll
            for (int e = 0; e < 8; ++e) {
                float v = acc[mi][ni][e] + bb;
                float g = 0.5f * v * (1.0f + erff(v * 0.70710678118654752440f));
                C[(size_t)(base_m + e) * N_DIM + col] = g;
            }
        }
    }
}

// ---------------------------------------------------------------------------
// Launch
// ---------------------------------------------------------------------------
extern "C" void kernel_launch(void* const* d_in, const int* in_sizes, int n_in,
                              void* d_out, int out_size, void* d_ws, size_t ws_size,
                              hipStream_t stream)
{
    const float* x     = (const float*)d_in[0];
    const float* gamma = (const float*)d_in[1];
    const float* beta  = (const float*)d_in[2];
    const float* W     = (const float*)d_in[3];
    const float* bias  = (const float*)d_in[4];
    float* out = (float*)d_out;

    // workspace: bf16 xn [M][K] then bf16 Wt [N][K]  (~30 MB total)
    __bf16* xn_bf = (__bf16*)d_ws;
    __bf16* wt_bf = (__bf16*)((char*)d_ws + (size_t)M_DIM * K_DIM * sizeof(__bf16));

    // 1) LayerNorm -> bf16 (one wave per row, 8 rows per block)
    ln_bf16_kernel<<<M_DIM / 8, 256, 0, stream>>>(x, gamma, beta, xn_bf);

    // 2) W fp32 -> bf16, transposed to [N][K]
    dim3 tgrid(K_DIM / 32, N_DIM / 32);  // (24, 96)
    w_transpose_bf16_kernel<<<tgrid, 256, 0, stream>>>(W, wt_bf);

    // 3) WMMA GEMM + bias + exact GELU
    dim3 grid(N_DIM / BN, M_DIM / BM);   // (24, 128)
    gemm_bias_gelu_kernel<<<grid, 256, 0, stream>>>(xn_bf, wt_bf, bias, out);
}
